// SoftTree_72275709657591
// MI455X (gfx1250) — matrix operand order repaired and checked by reading, not compiled, not run on hardware
//
#include <hip/hip_runtime.h>
#include <hip/hip_bf16.h>

// ---------------- SoftTree fused forward for gfx1250 (MI455X) ----------------
// out = leaf_probs(sigmoid(x@gw+gb)) @ z.T
//   x: (65536,512) f32   gw: (512,255) f32   gb: (255,) f32   z: (256,256) f32
//   out: (65536,256) f32
// Single fused main kernel: per-wave 16-row stripe, WMMA bf16 GEMM1 ->
// sigmoid -> tree expansion (LDS, aliased in place) -> WMMA bf16 GEMM2.
// No inter-wave communication => no barriers.

#define BATCH   65536
#define INF     512
#define OUTF    256
#define LEAVES  256
#define GATES   255
#define WAVES   8
#define ROWSTRIDE 264   // floats per LDS row (255 gates + pad; leaves overlay bytes 0..511)

typedef __attribute__((ext_vector_type(16))) __bf16 bf16x16;
typedef __attribute__((ext_vector_type(8)))  float  v8f;
typedef __attribute__((ext_vector_type(4)))  float  f32x4;

union FragU {
    bf16x16        b;
    f32x4          f[2];
    unsigned short u[16];
};

// fp32 -> bf16 round-to-nearest-even, as a bit pattern (avoids __bf16 arith)
static __device__ __forceinline__ unsigned short f2bf(float f) {
    unsigned int u = __float_as_uint(f);
    u += 0x7FFFu + ((u >> 16) & 1u);
    return (unsigned short)(u >> 16);
}

// ---------------- prep: gw (512x255 f32, K-major) -> B-fragment order -------
// gwB layout: [t=0..15][s=0..15][lane=0..31][e=0..15] bf16, 32B per lane chunk.
// B-matrix (KxN) per-lane map: n = t*16 + (lane&15); k = s*32 + 8*(lane>>4) + (e<8 ? e : e+8)
__global__ void prep_gwB(const float* __restrict__ gw, unsigned short* __restrict__ gwB) {
    int idx = blockIdx.x * 256 + threadIdx.x;           // 16*16*32 = 8192 threads
    if (idx >= 16 * 16 * 32) return;
    int lane = idx & 31;
    int s    = (idx >> 5) & 15;
    int t    = idx >> 9;
    int hi   = lane >> 4;
    int n    = t * 16 + (lane & 15);
#pragma unroll
    for (int e = 0; e < 16; ++e) {
        int   k = s * 32 + 8 * hi + (e < 8 ? e : e + 8);
        float v = (n < GATES) ? gw[k * GATES + n] : 0.0f;
        gwB[idx * 16 + e] = f2bf(v);
    }
}

// ---------------- prep: z (256x256 f32, [o][l]) -> B-fragment order ---------
// zB layout: [t=0..15][s=0..7][lane][e] bf16; B element (k=l, n=o) = z[n*256+k]
__global__ void prep_zB(const float* __restrict__ z, unsigned short* __restrict__ zB) {
    int idx = blockIdx.x * 256 + threadIdx.x;           // 16*8*32 = 4096 threads
    if (idx >= 16 * 8 * 32) return;
    int lane = idx & 31;
    int s    = (idx >> 5) & 7;
    int t    = idx >> 8;
    int hi   = lane >> 4;
    int n    = t * 16 + (lane & 15);
#pragma unroll
    for (int e = 0; e < 16; ++e) {
        int k = s * 32 + 8 * hi + (e < 8 ? e : e + 8);
        zB[idx * 16 + e] = f2bf(z[n * 256 + k]);
    }
}

// ---------------- fused main kernel ----------------------------------------
__global__ __launch_bounds__(256) void softtree_main(
    const float* __restrict__ x,
    const float* __restrict__ gb,
    const unsigned short* __restrict__ gwB,
    const unsigned short* __restrict__ zB,
    float* __restrict__ out)
{
    __shared__ float rowbuf[WAVES][16][ROWSTRIDE];

    const int wave = threadIdx.x >> 5;
    const int lane = threadIdx.x & 31;
    const int hi   = lane >> 4;        // half-wave select
    const int lm   = lane & 15;        // row (A) or col (B/C) within tile
    const int row0 = blockIdx.x * (WAVES * 16) + wave * 16;

    v8f acc[16];
#pragma unroll
    for (int t = 0; t < 16; ++t) acc[t] = (v8f){0.f,0.f,0.f,0.f,0.f,0.f,0.f,0.f};

    // ---- Stage 1: logits tile (16 x 256) = x_tile(16x512) @ gw(512x256pad) ----
    const float* xrow = x + (size_t)(row0 + lm) * INF;
#pragma unroll 1
    for (int s = 0; s < 16; ++s) {
        // A fragment: 16-bit A layout, k = s*32 + {0..7,16..23} + 8*hi
        const f32x4* p0 = (const f32x4*)(xrow + s * 32 + hi * 8);
        const f32x4* p1 = (const f32x4*)(xrow + s * 32 + 16 + hi * 8);
        f32x4 q[4];
        q[0] = p0[0]; q[1] = p0[1];
        q[2] = p1[0]; q[3] = p1[1];
        FragU af;
#pragma unroll
        for (int i = 0; i < 16; ++i) af.u[i] = f2bf(q[i >> 2][i & 3]);
#pragma unroll
        for (int t = 0; t < 16; ++t) {
            const f32x4* bp = (const f32x4*)(gwB + ((size_t)(t * 16 + s) * 32 + lane) * 16);
            FragU bf;
            bf.f[0] = bp[0];
            bf.f[1] = bp[1];
            acc[t] = __builtin_amdgcn_wmma_f32_16x16x32_bf16(
                false, af.b, false, bf.b, (short)0, acc[t], false, false);
        }
    }

    // ---- Stage 2: sigmoid(logit + gb) -> LDS (fp32 gates per row) ----
    asm volatile("" ::: "memory");
    float* my = &rowbuf[wave][0][0];
#pragma unroll 1
    for (int t = 0; t < 16; ++t) {
        int   n    = t * 16 + lm;
        float bias = (n < GATES) ? gb[n] : 0.0f;
#pragma unroll
        for (int r = 0; r < 8; ++r) {
            int   m  = r + hi * 8;
            float v  = acc[t][r] + bias;
            float gv = 1.0f / (1.0f + __expf(-v));
            my[m * ROWSTRIDE + n] = gv;
        }
    }
    asm volatile("" ::: "memory");

    // ---- Stage 3: tree expansion; lane computes leaves [lane*8, lane*8+8) ----
    // Leaves (bf16) overwrite bytes [0,512) of the same row: safe because all
    // gate reads of a row precede the store (DS ops are in-order per wave).
#pragma unroll 1
    for (int row = 0; row < 16; ++row) {
        float* gr = my + row * ROWSTRIDE;
        float  p  = 1.0f;
#pragma unroll
        for (int d = 0; d < 5; ++d) {
            int   node = (1 << d) - 1 + (lane >> (5 - d));
            int   bit  = (lane >> (4 - d)) & 1;
            float gv   = gr[node];
            p *= bit ? (1.0f - gv) : gv;
        }
        float g5  = gr[31 + lane];
        float g6a = gr[63 + 2 * lane];
        float g6b = gr[64 + 2 * lane];
        float g70 = gr[127 + 4 * lane];
        float g71 = gr[128 + 4 * lane];
        float g72 = gr[129 + 4 * lane];
        float g73 = gr[130 + 4 * lane];

        float q0  = p * g5;   float q1  = p  - q0;
        float r00 = q0 * g6a; float r01 = q0 - r00;
        float r10 = q1 * g6b; float r11 = q1 - r10;

        union { unsigned short h[8]; f32x4 f; } pk;
        float l0 = r00 * g70; pk.h[0] = f2bf(l0); pk.h[1] = f2bf(r00 - l0);
        float l2 = r01 * g71; pk.h[2] = f2bf(l2); pk.h[3] = f2bf(r01 - l2);
        float l4 = r10 * g72; pk.h[4] = f2bf(l4); pk.h[5] = f2bf(r10 - l4);
        float l6 = r11 * g73; pk.h[6] = f2bf(l6); pk.h[7] = f2bf(r11 - l6);

        ((f32x4*)gr)[lane] = pk.f;   // leaf cols [lane*8, lane*8+8) as bf16
    }
    asm volatile("" ::: "memory");

    // ---- Stage 4: out tile (16 x 256) = leaf_tile(16x256) @ z.T ----
#pragma unroll
    for (int t = 0; t < 16; ++t) acc[t] = (v8f){0.f,0.f,0.f,0.f,0.f,0.f,0.f,0.f};

    const char* myb = (const char*)my;
#pragma unroll 1
    for (int s = 0; s < 8; ++s) {
        FragU af;
        const f32x4* ap0 = (const f32x4*)(myb + (size_t)lm * ROWSTRIDE * 4 + (s * 32 + hi * 8) * 2);
        const f32x4* ap1 = (const f32x4*)(myb + (size_t)lm * ROWSTRIDE * 4 + (s * 32 + 16 + hi * 8) * 2);
        af.f[0] = ap0[0];
        af.f[1] = ap1[0];
#pragma unroll
        for (int t = 0; t < 16; ++t) {
            const f32x4* bp = (const f32x4*)(zB + ((size_t)(t * 8 + s) * 32 + lane) * 16);
            FragU bf;
            bf.f[0] = bp[0];
            bf.f[1] = bp[1];
            acc[t] = __builtin_amdgcn_wmma_f32_16x16x32_bf16(
                false, af.b, false, bf.b, (short)0, acc[t], false, false);
        }
    }

    // ---- store: C/D layout -> out[row0+m][t*16+lm] ----
#pragma unroll 1
    for (int t = 0; t < 16; ++t) {
#pragma unroll
        for (int r = 0; r < 8; ++r) {
            int m = r + hi * 8;
            out[(size_t)(row0 + m) * OUTF + t * 16 + lm] = acc[t][r];
        }
    }
}

// ---------------- host entry ----------------
extern "C" void kernel_launch(void* const* d_in, const int* in_sizes, int n_in,
                              void* d_out, int out_size, void* d_ws, size_t ws_size,
                              hipStream_t stream) {
    const float* x  = (const float*)d_in[0];
    const float* gw = (const float*)d_in[1];
    const float* gb = (const float*)d_in[2];
    const float* z  = (const float*)d_in[3];
    float* out = (float*)d_out;

    unsigned short* gwB = (unsigned short*)d_ws;                       // 256 KB
    unsigned short* zB  = (unsigned short*)((char*)d_ws + (size_t)16 * 16 * 32 * 16 * 2); // +256 KB

    prep_gwB<<<32, 256, 0, stream>>>(gw, gwB);
    prep_zB <<<16, 256, 0, stream>>>(z, zB);
    softtree_main<<<BATCH / (WAVES * 16), 256, 0, stream>>>(x, gb, gwB, zB, out);
}
